// WriteBondParameters_1623497638277
// MI455X (gfx1250) — matrix operand order repaired and checked by reading, not compiled, not run on hardware
//
#include <hip/hip_runtime.h>

// ---------------------------------------------------------------------------
// BondNet-style edge MLP, fused end-to-end per 32-edge tile.
// bf16 WMMA (v_wmma_f32_16x16x32_bf16), f32 accumulation, LN/ELU in f32.
// Normalized activations staged once per layer as bf16 in LDS; each wave
// computes 2 row-tiles x 4 col-tiles so every B fragment feeds 2 WMMAs.
// B fragments double-buffered. Residual layers run in rolled loops (one
// inlined body each) to keep static code inside the I-cache.
// ---------------------------------------------------------------------------

typedef __attribute__((ext_vector_type(16))) __bf16 v16bf;
typedef __attribute__((ext_vector_type(8)))  __bf16 v8bf;
typedef __attribute__((ext_vector_type(4)))  __bf16 v4bf;
typedef __attribute__((ext_vector_type(8)))  float  v8f;

#define TM       32          // edges per block
#define THREADS  256         // 8 wave32
#define LN_EPS   1e-5f

// packed-weight layout (elements):
//   stage0: K=1024 (32 ktiles) x 32 ntiles;  stages 1..5: K=512 (16 ktiles) x 32 ntiles
//   elem index = ((tk*32 + tn)*32 + lane)*16 + e
#define ST0_ELEMS (32*32*32*16)      // 524288
#define STN_ELEMS (16*32*32*16)      // 262144
#define PACK_TOTAL (ST0_ELEMS + 5*STN_ELEMS)

__global__ void pack_weights_kernel(const float* __restrict__ w0,
                                    const float* __restrict__ wsym,
                                    const float* __restrict__ wbond,
                                    __bf16* __restrict__ outw) {
  int idx = blockIdx.x * blockDim.x + threadIdx.x;
  if (idx >= PACK_TOTAL) return;
  const float* src;
  int local;
  if (idx < ST0_ELEMS) { src = w0; local = idx; }
  else {
    int r = idx - ST0_ELEMS;
    int s = r / STN_ELEMS;                       // 0..4
    local = r - s * STN_ELEMS;
    src = (s < 3) ? (wsym + (size_t)s * 512 * 512)
                  : (wbond + (size_t)(s - 3) * 512 * 512);
  }
  int e    = local & 15;
  int lane = (local >> 4) & 31;
  int tile = local >> 9;
  int tn   = tile & 31;
  int tk   = tile >> 5;
  int k    = tk * 32 + ((e & 8) ? 16 : 0) + ((lane & 16) ? 8 : 0) + (e & 7);
  int n    = tn * 16 + (lane & 15);
  outw[idx] = (__bf16)src[(size_t)k * 512 + n];
}

// branchless ELU: max(x,0) + exp(min(x,0)) - 1   (one v_exp_f32, no branches)
__device__ __forceinline__ float elu_f(float x) {
  return fmaxf(x, 0.f) + __expf(fminf(x, 0.f)) - 1.f;
}

// row stats (mean, rsqrt(var+eps)) for 32 rows of an LDS f32 buffer
__device__ __forceinline__ void block_stats_f32(const float* __restrict__ buf,
                                                float* red, float* stat,
                                                int t, int width) {
  int r = t & 31, seg = t >> 5;               // 8 segments per row
  int cper = width >> 3;
  float s1 = 0.f, s2 = 0.f;
  const float* row = buf + r * width + seg * cper;
  for (int c = 0; c < cper; c += 4) {
    float4 v = *(const float4*)(row + c);
    s1 += v.x + v.y + v.z + v.w;
    s2 += v.x * v.x + v.y * v.y + v.z * v.z + v.w * v.w;
  }
  red[(r * 8 + seg) * 2]     = s1;
  red[(r * 8 + seg) * 2 + 1] = s2;
  __syncthreads();
  if (t < 32) {
    float a = 0.f, b = 0.f;
    for (int sgi = 0; sgi < 8; ++sgi) {
      a += red[(t * 8 + sgi) * 2];
      b += red[(t * 8 + sgi) * 2 + 1];
    }
    float inv = 1.f / (float)width;
    float mu = a * inv;
    float var = b * inv - mu * mu;
    stat[2 * t]     = mu;
    stat[2 * t + 1] = rsqrtf(var + LN_EPS);
  }
  __syncthreads();
}

__device__ __forceinline__ void block_stats_bf16(const __bf16* __restrict__ buf,
                                                 float* red, float* stat, int t) {
  int r = t & 31, seg = t >> 5;               // width 1024 -> 128 per segment
  float s1 = 0.f, s2 = 0.f;
  const __bf16* row = buf + r * 1024 + seg * 128;
  for (int c = 0; c < 128; c += 8) {
    v8bf v = *(const v8bf*)(row + c);
#pragma unroll
    for (int u = 0; u < 8; ++u) { float f = (float)v[u]; s1 += f; s2 += f * f; }
  }
  red[(r * 8 + seg) * 2]     = s1;
  red[(r * 8 + seg) * 2 + 1] = s2;
  __syncthreads();
  if (t < 32) {
    float a = 0.f, b = 0.f;
    for (int sgi = 0; sgi < 8; ++sgi) {
      a += red[(t * 8 + sgi) * 2];
      b += red[(t * 8 + sgi) * 2 + 1];
    }
    float mu = a * (1.f / 1024.f);
    float var = b * (1.f / 1024.f) - mu * mu;
    stat[2 * t]     = mu;
    stat[2 * t + 1] = rsqrtf(var + LN_EPS);
  }
  __syncthreads();
}

// A fragment from row-major bf16 LDS tile: two contiguous 8-element runs
__device__ __forceinline__ v16bf load_afrag(const __bf16* __restrict__ arow,
                                            int tk, int hiA) {
  const __bf16* abase = arow + tk * 32 + hiA;
  v8bf lo = *(const v8bf*)(abase);        // k0 .. k0+7
  v8bf hi = *(const v8bf*)(abase + 16);   // k0+16 .. k0+23
  return __builtin_shufflevector(lo, hi, 0, 1, 2, 3, 4, 5, 6, 7,
                                         8, 9, 10, 11, 12, 13, 14, 15);
}

// K-loop, software-pipelined B fragments shared by 2 row-tiles (2 WMMAs / B).
// acc[0..3] = row-tile 0 x 4 col-tiles, acc[4..7] = row-tile 1 x 4 col-tiles.
template <int KT, int WIDTH>
__device__ __forceinline__ void gemm_tiles(const __bf16* __restrict__ xnb,
                                           const __bf16* __restrict__ wp,
                                           int wave, int lane, v8f* acc) {
  int hiA = (lane & 16) ? 8 : 0;
  const __bf16* arow0 = xnb + (lane & 15) * WIDTH;
  const __bf16* arow1 = xnb + (16 + (lane & 15)) * WIDTH;
  const __bf16* bp0 = wp + ((size_t)(wave * 4) * 32 + lane) * 16;
  v16bf bc[4];
#pragma unroll
  for (int q = 0; q < 4; ++q) bc[q] = *(const v16bf*)(bp0 + (size_t)q * 512);
#pragma unroll 2
  for (int tk = 0; tk < KT; ++tk) {
    v16bf bn[4];
    if (tk + 1 < KT) {
      const __bf16* bpn = bp0 + (size_t)(tk + 1) * (32 * 32 * 16);
#pragma unroll
      for (int q = 0; q < 4; ++q) bn[q] = *(const v16bf*)(bpn + (size_t)q * 512);
    }
    v16bf a0 = load_afrag(arow0, tk, hiA);
    v16bf a1 = load_afrag(arow1, tk, hiA);
#pragma unroll
    for (int q = 0; q < 4; ++q) {
      acc[q] = __builtin_amdgcn_wmma_f32_16x16x32_bf16(
          false, a0, false, bc[q], (short)0, acc[q], false, false);
    }
#pragma unroll
    for (int q = 0; q < 4; ++q) {
      acc[4 + q] = __builtin_amdgcn_wmma_f32_16x16x32_bf16(
          false, a1, false, bc[q], (short)0, acc[4 + q], false, false);
    }
#pragma unroll
    for (int q = 0; q < 4; ++q) bc[q] = bn[q];
  }
}

// residual block (in place): act = elu(LN(act) @ W + b) + LN(act)  (32x512, K=512)
__device__ __forceinline__ void skipped_layer(
    float* __restrict__ act,
    const float* __restrict__ lnwg, const float* __restrict__ lnbg,
    const float* __restrict__ bias, const __bf16* __restrict__ wp,
    __bf16* xnb, float* lnw_s, float* lnb_s, float* red, float* stat, int t) {
  for (int i = t; i < 512; i += THREADS) { lnw_s[i] = lnwg[i]; lnb_s[i] = lnbg[i]; }
  __syncthreads();
  block_stats_f32(act, red, stat, t, 512);

  // stage xn = LN(act) as bf16, row-major (each element computed once)
  {
    int r = t >> 3, c8 = t & 7;
    float mu = stat[2 * r], rs = stat[2 * r + 1];
    const float* row = act + r * 512;
    __bf16* dst = xnb + r * 512;
#pragma unroll 4
    for (int j = 0; j < 16; ++j) {
      int col = c8 * 64 + j * 4;
      float4 v = *(const float4*)(row + col);
      float4 w = *(const float4*)(lnw_s + col);
      float4 b = *(const float4*)(lnb_s + col);
      v4bf pk = { (__bf16)((v.x - mu) * rs * w.x + b.x),
                  (__bf16)((v.y - mu) * rs * w.y + b.y),
                  (__bf16)((v.z - mu) * rs * w.z + b.z),
                  (__bf16)((v.w - mu) * rs * w.w + b.w) };
      *(v4bf*)(dst + col) = pk;
    }
  }
  __syncthreads();

  int wave = t >> 5, lane = t & 31;
  int hiA = (lane & 16) ? 8 : 0;

  v8f acc[8] = {};
  gemm_tiles<16, 512>(xnb, wp, wave, lane, acc);

#pragma unroll
  for (int rt = 0; rt < 2; ++rt) {
#pragma unroll
    for (int q = 0; q < 4; ++q) {
      int n0 = (wave * 4 + q) * 16 + (lane & 15);
      float bv = bias[n0];
#pragma unroll
      for (int j = 0; j < 8; ++j) {
        int mo = rt * 16 + j + hiA;
        float z  = acc[rt * 4 + q][j] + bv;
        float xn = (float)xnb[mo * 512 + n0];
        act[mo * 512 + n0] = elu_f(z) + xn;
      }
    }
  }
  __syncthreads();
}

// first symmetric layer: act = elu(LN(concat_p) @ W0 + b0)   (K=1024)
__device__ __forceinline__ void sym0_layer(
    int p, const __bf16* __restrict__ xin, float* __restrict__ act,
    const float* __restrict__ lnwg, const float* __restrict__ lnbg,
    const float* __restrict__ bias, const __bf16* __restrict__ wp,
    __bf16* xnb, float* lnw_s, float* lnb_s, float* red, float* stat, int t) {
  for (int i = t; i < 1024; i += THREADS) { lnw_s[i] = lnwg[i]; lnb_s[i] = lnbg[i]; }
  __syncthreads();
  block_stats_bf16(xin, red, stat, t);      // stats are permutation-invariant

  // stage xn_p (swapped ordering == column permutation of the same tile)
  {
    int r = t >> 3, c8 = t & 7;
    float mu = stat[2 * r], rs = stat[2 * r + 1];
    const __bf16* row = xin + r * 1024;
    __bf16* dst = xnb + r * 1024;
#pragma unroll 4
    for (int j = 0; j < 32; ++j) {
      int k = c8 * 128 + j * 4;
      int c = p ? ((k + 512) & 1023) : k;
      v4bf xv = *(const v4bf*)(row + c);
      float4 w = *(const float4*)(lnw_s + k);
      float4 b = *(const float4*)(lnb_s + k);
      v4bf pk = { (__bf16)(((float)xv[0] - mu) * rs * w.x + b.x),
                  (__bf16)(((float)xv[1] - mu) * rs * w.y + b.y),
                  (__bf16)(((float)xv[2] - mu) * rs * w.z + b.z),
                  (__bf16)(((float)xv[3] - mu) * rs * w.w + b.w) };
      *(v4bf*)(dst + k) = pk;
    }
  }
  __syncthreads();

  int wave = t >> 5, lane = t & 31;
  int hiA = (lane & 16) ? 8 : 0;

  v8f acc[8] = {};
  gemm_tiles<32, 1024>(xnb, wp, wave, lane, acc);

#pragma unroll
  for (int rt = 0; rt < 2; ++rt) {
#pragma unroll
    for (int q = 0; q < 4; ++q) {
      int n0 = (wave * 4 + q) * 16 + (lane & 15);
      float bv = bias[n0];
#pragma unroll
      for (int j = 0; j < 8; ++j) {
        int mo = rt * 16 + j + hiA;
        act[mo * 512 + n0] = elu_f(acc[rt * 4 + q][j] + bv);
      }
    }
  }
  __syncthreads();
}

// LDS (266 KB) limits us to 1 block/WGP = 2 waves/SIMD, so allow up to
// 512 VGPRs per wave (min 2 waves per SIMD) -> no scratch spills.
__global__ void __launch_bounds__(THREADS, 2) bondnet_kernel(
    const int* __restrict__ pairs, const float* __restrict__ h,
    const float* __restrict__ sym0_lnw, const float* __restrict__ sym0_lnb,
    const float* __restrict__ sym0_b,
    const float* __restrict__ sym_lnw, const float* __restrict__ sym_lnb,
    const float* __restrict__ sym_b,
    const float* __restrict__ bond_lnw, const float* __restrict__ bond_lnb,
    const float* __restrict__ bond_b,
    const float* __restrict__ out_W, const float* __restrict__ out_b,
    const __bf16* __restrict__ wpacked, float* __restrict__ outp) {
  extern __shared__ char smem[];
  __bf16* xin  = (__bf16*)smem;                   // 32x1024 bf16   64 KB
  __bf16* xnb  = xin + TM * 1024;                 // 32x1024 bf16   64 KB
  float*  act  = (float*)(xnb + TM * 1024);       // 32x512  f32    64 KB
  float*  sbuf = act + TM * 512;                  //                64 KB
  float*  lnw_s = sbuf + TM * 512;                // 1024 f32        4 KB
  float*  lnb_s = lnw_s + 1024;                   //                 4 KB
  float*  red   = lnb_s + 1024;                   // 512 f32         2 KB
  float*  stat  = red + 512;                      // 64 f32

  const int t  = threadIdx.x;
  const int e0 = blockIdx.x * TM;

  // ---- gather h[pairs] -> LDS (bf16), 8 threads per edge row ----
  {
    int r = t >> 3, c8 = t & 7;
    int i0 = pairs[(size_t)(e0 + r) * 2 + 0];
    int i1 = pairs[(size_t)(e0 + r) * 2 + 1];
    const float* h0 = h + (size_t)i0 * 512;
    const float* h1 = h + (size_t)i1 * 512;
#pragma unroll 4
    for (int j = 0; j < 16; ++j) {
      int col = c8 * 64 + j * 4;
      float4 v0 = *(const float4*)(h0 + col);
      float4 v1 = *(const float4*)(h1 + col);
      v4bf p0 = { (__bf16)v0.x, (__bf16)v0.y, (__bf16)v0.z, (__bf16)v0.w };
      v4bf p1 = { (__bf16)v1.x, (__bf16)v1.y, (__bf16)v1.z, (__bf16)v1.w };
      *(v4bf*)(xin + r * 1024 + col)       = p0;
      *(v4bf*)(xin + r * 1024 + 512 + col) = p1;
    }
  }
  __syncthreads();

  const __bf16* wp0 = wpacked;
  const __bf16* wps = wpacked + ST0_ELEMS;

  // ---- s = sym([f1,f2]) + sym([f2,f1]); p=0 -> sbuf=act, p=1 -> act+=sbuf ----
#pragma clang loop unroll(disable)
  for (int p = 0; p < 2; ++p) {
    sym0_layer(p, xin, act, sym0_lnw, sym0_lnb, sym0_b, wp0,
               xnb, lnw_s, lnb_s, red, stat, t);
#pragma clang loop unroll(disable)
    for (int l = 0; l < 3; ++l) {
      skipped_layer(act, sym_lnw + l * 512, sym_lnb + l * 512, sym_b + l * 512,
                    wps + (size_t)l * STN_ELEMS,
                    xnb, lnw_s, lnb_s, red, stat, t);
    }
    for (int i = t; i < TM * 512; i += THREADS) {
      if (p == 0) sbuf[i] = act[i]; else act[i] += sbuf[i];
    }
    __syncthreads();
  }

  // ---- bond residual blocks (in place on act) ----
#pragma clang loop unroll(disable)
  for (int l = 0; l < 2; ++l) {
    skipped_layer(act, bond_lnw + l * 512, bond_lnb + l * 512, bond_b + l * 512,
                  wps + (size_t)(3 + l) * STN_ELEMS,
                  xnb, lnw_s, lnb_s, red, stat, t);
  }

  // ---- head: coeffs = s @ out_W + out_b; softplus transforms ----
  {
    int r = t >> 3, seg = t & 7;
    const float* row = act + r * 512 + seg * 64;
    float p0 = 0.f, p1 = 0.f;
    for (int c = 0; c < 64; ++c) {
      float v = row[c];
      float2 w = *(const float2*)(out_W + (size_t)(seg * 64 + c) * 2);
      p0 += v * w.x;
      p1 += v * w.y;
    }
    red[(r * 8 + seg) * 2]     = p0;
    red[(r * 8 + seg) * 2 + 1] = p1;
  }
  __syncthreads();
  if (t < 64) {
    int m = t >> 1, o = t & 1;
    float s = 0.f;
    for (int seg = 0; seg < 8; ++seg) s += red[(m * 8 + seg) * 2 + o];
    s += out_b[o];
    // stable softplus: max(x,0) + log(1 + exp(-|x|))
    float x  = s + (o == 0 ? 7.5f : 2.0f);      // MEAN/STD shifts
    float sp = fmaxf(x, 0.f) + __logf(1.f + __expf(-fabsf(x)));
    float res = (o == 0 ? 0.2f : 300.f) * sp;   // STD scales
    outp[(size_t)(e0 + m) * 2 + o] = res;
  }
}

#define SMEM_BYTES (2 * 65536 /*xin,xnb*/ + 2 * 65536 /*act,sbuf*/ \
                    + 8192 /*ln*/ + 2048 /*red*/ + 256 /*stat*/)

extern "C" void kernel_launch(void* const* d_in, const int* in_sizes, int n_in,
                              void* d_out, int out_size, void* d_ws, size_t ws_size,
                              hipStream_t stream) {
  const int*   pairs    = (const int*)d_in[0];
  const float* h        = (const float*)d_in[1];
  const float* sym0_lnw = (const float*)d_in[2];
  const float* sym0_lnb = (const float*)d_in[3];
  const float* sym0_W   = (const float*)d_in[4];
  const float* sym0_b   = (const float*)d_in[5];
  const float* sym_lnw  = (const float*)d_in[6];
  const float* sym_lnb  = (const float*)d_in[7];
  const float* sym_W    = (const float*)d_in[8];
  const float* sym_b    = (const float*)d_in[9];
  const float* bond_lnw = (const float*)d_in[10];
  const float* bond_lnb = (const float*)d_in[11];
  const float* bond_W   = (const float*)d_in[12];
  const float* bond_b   = (const float*)d_in[13];
  const float* out_W    = (const float*)d_in[14];
  const float* out_b    = (const float*)d_in[15];
  float* outp = (float*)d_out;

  __bf16* wpacked = (__bf16*)d_ws;   // needs PACK_TOTAL*2 = 3.67 MB of scratch

  pack_weights_kernel<<<(PACK_TOTAL + 255) / 256, 256, 0, stream>>>(
      sym0_W, sym_W, bond_W, wpacked);

  int E = in_sizes[0] / 2;
  int nblocks = E / TM;
  bondnet_kernel<<<nblocks, THREADS, SMEM_BYTES, stream>>>(
      pairs, h, sym0_lnw, sym0_lnb, sym0_b, sym_lnw, sym_lnb, sym_b,
      bond_lnw, bond_lnb, bond_b, out_W, out_b, wpacked, outp);
}
